// SingleHeadAttention_78469052498069
// MI455X (gfx1250) — matrix operand order repaired and checked by reading, not compiled, not run on hardware
//
#include <hip/hip_runtime.h>
#include <hip/hip_bf16.h>

// Problem constants (match reference: B=4, T=4096, D=256)
#define BATCH 4
#define TSEQ  4096
#define DDIM  256

typedef _Float16 v16h __attribute__((ext_vector_type(16)));
typedef float    v8f  __attribute__((ext_vector_type(8)));
typedef unsigned int u32x4 __attribute__((ext_vector_type(4)));
typedef int          i32x8 __attribute__((ext_vector_type(8)));
typedef int          i32x4 __attribute__((ext_vector_type(4)));

#define WMMA_F16(a, b, c) \
  __builtin_amdgcn_wmma_f32_16x16x32_f16(false, (a), false, (b), (short)0, (c), false, false)

#if defined(__gfx1250__) && __has_builtin(__builtin_amdgcn_tensor_load_to_lds)
#define USE_TDM 1
#else
#define USE_TDM 0
#endif

__device__ __forceinline__ v8f zero_v8f() {
  v8f z;
#pragma unroll
  for (int j = 0; j < 8; ++j) z[j] = 0.f;
  return z;
}

__device__ __forceinline__ void wait_tensorcnt0() {
#if defined(__gfx1250__)
#if __has_builtin(__builtin_amdgcn_s_wait_tensorcnt)
  __builtin_amdgcn_s_wait_tensorcnt(0);
#else
  asm volatile("s_wait_tensorcnt 0x0" ::: "memory");
#endif
#endif
}

#if USE_TDM
// Issue a 2-D TDM tile load: global (2-byte elements, row length tensor_x,
// row stride stride_x) -> LDS (tile_x * tile_y elements, row-major, packed).
// D# layout per CDNA5 ISA ch.8 (group0 128b, group1 256b; groups 2/3 zero).
// This toolchain exposes the 6-arg builtin (extra int32x8 group before cpol).
__device__ __forceinline__ void tdm_load_2d(const void* gptr, unsigned lds_off,
                                            unsigned tile_x, unsigned tile_y,
                                            unsigned tensor_x, unsigned tensor_y,
                                            unsigned stride_x) {
  const unsigned long long ga =
      (unsigned long long)(uintptr_t)gptr & 0x01FFFFFFFFFFFFFFull;
  u32x4 g0;
  g0[0] = 1u;                                   // count=1, user descriptor
  g0[1] = lds_off;                              // LDS byte address
  g0[2] = (unsigned)ga;                         // global_addr[31:0]
  g0[3] = (unsigned)(ga >> 32) | (2u << 30);    // global_addr[56:32] | type=2
  i32x8 g1;
  g1[0] = (int)(1u << 16);                      // wg_mask=0, data_size=1 (2B)
  g1[1] = (int)((tensor_x & 0xffffu) << 16);    // tensor_dim0[15:0]
  g1[2] = (int)((tensor_x >> 16) | ((tensor_y & 0xffffu) << 16));
  g1[3] = (int)((tensor_y >> 16) | (tile_x << 16));   // tile_dim0
  g1[4] = (int)(tile_y & 0xffffu);              // tile_dim1 (tile_dim2=0)
  g1[5] = (int)stride_x;                        // tensor_dim0_stride[31:0]
  g1[6] = 0;                                    // stride0 hi | stride1 lo
  g1[7] = 0;                                    // stride1 hi
  const i32x4 z4 = {0, 0, 0, 0};
  const i32x8 z8 = {0, 0, 0, 0, 0, 0, 0, 0};
  __builtin_amdgcn_tensor_load_to_lds(g0, g1, z4, z4, z8, 0);
}
#endif

// ---- Fragment loaders -------------------------------------------------------
// A-frag (16x32, 16-bit): lanes 0-15 row m=lane hold K=0..15 contiguous;
// lanes 16-31 row m=lane-16 hold K=16..31.
__device__ __forceinline__ v16h fragA_h(const _Float16* p, int row, int stride,
                                        int kbase, int lane) {
  const int lo = (lane & 16) ? 16 : 0;
  const _Float16* r = p + (size_t)row * stride + kbase + lo;
  v16h f;
#pragma unroll
  for (int j = 0; j < 16; ++j) f[j] = r[j];
  return f;
}

__device__ __forceinline__ v16h fragA_f32(const float* p, int row, int stride,
                                          int kbase, int lane) {
  const int lo = (lane & 16) ? 16 : 0;
  const float* r = p + (size_t)row * stride + kbase + lo;
  v16h f;
#pragma unroll
  for (int j = 0; j < 16; ++j) f[j] = (_Float16)r[j];
  return f;
}

// B-frag (32x16, 16-bit): lanes 0-15 (n=lane) hold K=0..15 contiguous;
// lanes 16-31 (n=lane-16) hold K=16..31.
__device__ __forceinline__ v16h fragB_h(const _Float16* p, int row, int stride,
                                        int kbase, int lane) {
  const int off = (lane & 16) ? 16 : 0;
  const _Float16* r = p + (size_t)row * stride + kbase + off;
  v16h f;
#pragma unroll
  for (int j = 0; j < 16; ++j) f[j] = r[j];
  return f;
}

__device__ __forceinline__ v16h fragB_f32(const float* p, int row, int stride,
                                          int kbase, int lane) {
  const int off = (lane & 16) ? 16 : 0;
  const float* r = p + (size_t)row * stride + kbase + off;
  v16h f;
#pragma unroll
  for (int j = 0; j < 16; ++j) f[j] = (_Float16)r[j];
  return f;
}

// ---- Kernel 1: Q/K projections, fp32 -> f16, row-major [b][t][e] ------------
__global__ void __launch_bounds__(32, 1)
qk_proj_kernel(const float* __restrict__ x, const float* __restrict__ Wq,
               const float* __restrict__ Wk, _Float16* __restrict__ Qh,
               _Float16* __restrict__ Kh) {
  const int lane = threadIdx.x & 31;
  const int blk = blockIdx.x;          // BATCH * (TSEQ/16)
  const int b = blk >> 8;
  const int tt = blk & 255;
  const float* xb = x + (size_t)b * TSEQ * DDIM;

  v16h a[8];
#pragma unroll
  for (int c = 0; c < 8; ++c)
    a[c] = fragA_f32(xb, tt * 16 + (lane & 15), DDIM, c * 32, lane);

  _Float16* Qb = Qh + (size_t)b * TSEQ * DDIM;
  _Float16* Kb = Kh + (size_t)b * TSEQ * DDIM;
  const int hi8 = (lane & 16) ? 8 : 0;

  for (int n = 0; n < 16; ++n) {
    v8f cq = zero_v8f();
    v8f ck = zero_v8f();
#pragma unroll
    for (int c = 0; c < 8; ++c) {
      v16h bw = fragB_f32(Wq, n * 16 + (lane & 15), DDIM, c * 32, lane);
      cq = WMMA_F16(a[c], bw, cq);
      v16h bw2 = fragB_f32(Wk, n * 16 + (lane & 15), DDIM, c * 32, lane);
      ck = WMMA_F16(a[c], bw2, ck);
    }
    const int e = n * 16 + (lane & 15);
#pragma unroll
    for (int r = 0; r < 8; ++r) {
      const size_t idx = (size_t)(tt * 16 + hi8 + r) * DDIM + e;
      Qb[idx] = (_Float16)cq[r];
      Kb[idx] = (_Float16)ck[r];
    }
  }
}

// ---- Kernel 2: V projection, stored transposed: Vt[b][v][t] (f16) -----------
__global__ void __launch_bounds__(32, 1)
vt_proj_kernel(const float* __restrict__ x, const float* __restrict__ Wv,
               _Float16* __restrict__ Vth) {
  const int lane = threadIdx.x & 31;
  const int blk = blockIdx.x;          // BATCH * 16 * 256
  const int b = blk >> 12;
  const int vt = (blk >> 8) & 15;
  const int tt = blk & 255;
  const float* xb = x + (size_t)b * TSEQ * DDIM;

  v8f c = zero_v8f();
#pragma unroll
  for (int cc = 0; cc < 8; ++cc) {
    v16h aw = fragA_f32(Wv, vt * 16 + (lane & 15), DDIM, cc * 32, lane);
    v16h bx = fragB_f32(xb, tt * 16 + (lane & 15), DDIM, cc * 32, lane);
    c = WMMA_F16(aw, bx, c);
  }

  _Float16* Vb = Vth + (size_t)b * DDIM * TSEQ;
  const int t = tt * 16 + (lane & 15);
  const int hi8 = (lane & 16) ? 8 : 0;
#pragma unroll
  for (int r = 0; r < 8; ++r) {
    const int v = vt * 16 + hi8 + r;
    Vb[(size_t)v * TSEQ + t] = (_Float16)c[r];
  }
}

// ---- Kernel 3: causal flash attention ---------------------------------------
// 4 waves / block share a 64-row q block.  K and Vᵀ 32-key tiles are staged
// into LDS by the Tensor Data Mover (double-buffered, TENSORcnt-synced);
// each wave runs Sᵀ = K·Qᵀ, per-lane online softmax, outᵀ = Vᵀ·Pᵀ.
__global__ void __launch_bounds__(128, 1)
attn_kernel(const _Float16* __restrict__ Qh, const _Float16* __restrict__ Kh,
            const _Float16* __restrict__ Vth, float* __restrict__ out) {
  __shared__ __align__(16) _Float16 lk[2][32 * DDIM];   // K tile: [row][d]
  __shared__ __align__(16) _Float16 lv[2][DDIM * 32];   // Vᵀ tile: [v][k]

  const int lane = threadIdx.x & 31;
  const int wave = threadIdx.x >> 5;
  const int b = blockIdx.x >> 6;                  // TSEQ/64 blocks per batch
  const int q0blk = (blockIdx.x & 63) << 6;       // 64-row q block
  const int q0 = q0blk + wave * 16;               // this wave's q tile

  const _Float16* Qb = Qh + (size_t)b * TSEQ * DDIM;
  const _Float16* Kb = Kh + (size_t)b * TSEQ * DDIM;
  const _Float16* Vb = Vth + (size_t)b * DDIM * TSEQ;

  // Resident Q fragments (B operand of the score WMMAs)
  v16h bq[8];
#pragma unroll
  for (int c = 0; c < 8; ++c)
    bq[c] = fragB_h(Qb, q0 + (lane & 15), DDIM, c * 32, lane);

  v8f acc[16];
#pragma unroll
  for (int f = 0; f < 16; ++f) acc[f] = zero_v8f();

  float m_run = -__builtin_inff();
  float l_run = 0.f;

  const float inv = 0.0625f;                       // 1/sqrt(256)
  const int q_lane = q0 + (lane & 15);
  const int hi8 = (lane & 16) ? 8 : 0;

  // Shared loop: cover all k < q0blk+64.  Block 0 sweeps the whole sequence
  // (the reference's mask adds the SAME -1e10 to every logit of row 0 ->
  // uniform softmax over ALL keys).  Extra masked steps exp() to exactly 0.
  const int nsteps = (q0blk == 0) ? (TSEQ / 32) : ((q0blk >> 5) + 2);

#if USE_TDM
  if (wave == 0) {
    tdm_load_2d(Kb, (unsigned)(uintptr_t)&lk[0][0], DDIM, 32, DDIM, TSEQ, DDIM);
    tdm_load_2d(Vb, (unsigned)(uintptr_t)&lv[0][0], 32, DDIM, TSEQ, DDIM, TSEQ);
  }
#endif

  for (int s = 0; s < nsteps; ++s) {
    const int kbase = s << 5;
#if USE_TDM
    const int cur = s & 1;
    if (wave == 0) wait_tensorcnt0();   // step-s tiles landed in LDS
    __syncthreads();                    // visible to all; prev buffer now free
    if (wave == 0 && s + 1 < nsteps) {  // prefetch next tiles (overlaps compute)
      const int kn = kbase + 32;
      tdm_load_2d(Kb + (size_t)kn * DDIM, (unsigned)(uintptr_t)&lk[cur ^ 1][0],
                  DDIM, 32, DDIM, TSEQ, DDIM);
      tdm_load_2d(Vb + kn, (unsigned)(uintptr_t)&lv[cur ^ 1][0],
                  32, DDIM, TSEQ, DDIM, TSEQ);
    }
#else
    const int cur = 0;
    __syncthreads();                    // previous iteration done reading
    {                                   // cooperative staged copy
      const uint4* gk = (const uint4*)(Kb + (size_t)kbase * DDIM);
      uint4* sk = (uint4*)&lk[0][0];
      for (int i = threadIdx.x; i < (32 * DDIM) / 8; i += 128) sk[i] = gk[i];
      uint4* sv = (uint4*)&lv[0][0];
      for (int i = threadIdx.x; i < DDIM * 4; i += 128) {
        const int row = i >> 2, c4 = i & 3;
        sv[i] = *(const uint4*)(Vb + (size_t)row * TSEQ + kbase + c4 * 8);
      }
    }
    __syncthreads();
#endif
    const _Float16* lkc = &lk[cur][0];
    const _Float16* lvc = &lv[cur][0];

    // Sᵀ tiles from LDS: rows k (16 each), cols q (16).
    v8f st0 = zero_v8f();
    v8f st1 = zero_v8f();
#pragma unroll
    for (int c = 0; c < 8; ++c) {
      v16h a0 = fragA_h(lkc, (lane & 15), DDIM, c * 32, lane);
      v16h a1 = fragA_h(lkc, 16 + (lane & 15), DDIM, c * 32, lane);
      st0 = WMMA_F16(a0, bq[c], st0);
      st1 = WMMA_F16(a1, bq[c], st1);
    }

    // Mask (added BEFORE 1/sqrt(d) scale, like the reference) + logits.
    float z0[8], z1[8];
    float tmax = -__builtin_inff();
#pragma unroll
    for (int r = 0; r < 8; ++r) {
      const int k0 = kbase + r + hi8;
      const int k1 = k0 + 16;
      float s0 = st0[r] + ((k0 >= q_lane) ? -1e10f : 0.f);
      float s1 = st1[r] + ((k1 >= q_lane) ? -1e10f : 0.f);
      z0[r] = s0 * inv;
      z1[r] = s1 * inv;
      tmax = fmaxf(tmax, fmaxf(z0[r], z1[r]));
    }
    tmax = fmaxf(tmax, __shfl_xor(tmax, 16, 32));

    const float m_new = fmaxf(m_run, tmax);
    const float alpha = __expf(m_run - m_new);
    float p0[8], p1[8];
    float tsum = 0.f;
#pragma unroll
    for (int r = 0; r < 8; ++r) {
      p0[r] = __expf(z0[r] - m_new);
      p1[r] = __expf(z1[r] - m_new);
      tsum += p0[r] + p1[r];
    }
    tsum += __shfl_xor(tsum, 16, 32);
    m_run = m_new;
    l_run = l_run * alpha + tsum;

    // Build Pᵀ B-fragment (K = key index, N = q).
    v16h bp;
#pragma unroll
    for (int r = 0; r < 8; ++r) {
      const float o0 = __shfl_xor(p0[r], 16, 32);
      const float o1 = __shfl_xor(p1[r], 16, 32);
      const float lo = (lane & 16) ? o1 : p0[r];
      const float hi = (lane & 16) ? p1[r] : o0;
      bp[r] = (_Float16)lo;
      bp[8 + r] = (_Float16)hi;
    }

    // Rescale and accumulate outᵀ = Vᵀ · Pᵀ over all 256 v rows (from LDS).
#pragma unroll
    for (int f = 0; f < 16; ++f) {
#pragma unroll
      for (int r = 0; r < 8; ++r) acc[f][r] *= alpha;
      v16h av = fragA_h(lvc, f * 16 + (lane & 15), 32, 0, lane);
      acc[f] = WMMA_F16(av, bp, acc[f]);
    }
  }

  // Epilogue: out[b][q][v] = accᵀ / l.  Lane owns column q = lane&15.
  const float invl = 1.f / l_run;
  float* ob = out + ((size_t)b * TSEQ + q0 + (lane & 15)) * DDIM;
#pragma unroll
  for (int f = 0; f < 16; ++f) {
    float4 w0 = make_float4(acc[f][0] * invl, acc[f][1] * invl,
                            acc[f][2] * invl, acc[f][3] * invl);
    float4 w1 = make_float4(acc[f][4] * invl, acc[f][5] * invl,
                            acc[f][6] * invl, acc[f][7] * invl);
    *(float4*)(ob + f * 16 + hi8) = w0;
    *(float4*)(ob + f * 16 + hi8 + 4) = w1;
  }
}

// ---- Launch -----------------------------------------------------------------
extern "C" void kernel_launch(void* const* d_in, const int* in_sizes, int n_in,
                              void* d_out, int out_size, void* d_ws,
                              size_t ws_size, hipStream_t stream) {
  const float* x  = (const float*)d_in[0];
  const float* Wq = (const float*)d_in[1];
  const float* Wk = (const float*)d_in[2];
  const float* Wv = (const float*)d_in[3];
  // d_in[4] = causal_mask: computed analytically in-kernel.
  float* out = (float*)d_out;

  const size_t nQ = (size_t)BATCH * TSEQ * DDIM;   // elements per tensor
  _Float16* Qh  = (_Float16*)d_ws;                 // ws usage: 24 MB
  _Float16* Kh  = Qh + nQ;
  _Float16* Vth = Kh + nQ;

  qk_proj_kernel<<<BATCH * (TSEQ / 16), 32, 0, stream>>>(x, Wq, Wk, Qh, Kh);
  vt_proj_kernel<<<BATCH * (DDIM / 16) * (TSEQ / 16), 32, 0, stream>>>(x, Wv, Vth);
  attn_kernel<<<BATCH * (TSEQ / 64), 128, 0, stream>>>(Qh, Kh, Vth, out);
}
